// HeadAttention_51694226375393
// MI455X (gfx1250) — matrix operand order repaired and checked by reading, not compiled, not run on hardware
//
#include <hip/hip_runtime.h>
#include <hip/hip_bf16.h>
#include <math.h>

// ---------------------------------------------------------------------------
// HeadAttention (B=8, S=2048, EMB=2048, HEAD=128) for MI455X / gfx1250.
// bf16 WMMA (v_wmma_f32_16x16x32_bf16) everywhere, f32 accumulation.
// ---------------------------------------------------------------------------

typedef __attribute__((ext_vector_type(16))) __bf16 v16bf;
typedef __attribute__((ext_vector_type(8)))  __bf16 v8bf;   // 16B = one b128
typedef __attribute__((ext_vector_type(8)))  float  v8f;
typedef __attribute__((ext_vector_type(4)))  float  v4f;

#define B_    8
#define S_    2048
#define EMB_  2048
#define HEAD_ 128

static __device__ __forceinline__ v16bf join16(v8bf lo, v8bf hi) {
  v16bf r;
#pragma unroll
  for (int i = 0; i < 8; ++i) { r[i] = lo[i]; r[i + 8] = hi[i]; }
  return r;
}

// ---------------------------------------------------------------------------
// Kernel 0: Wt[mat][n][k] = (bf16) W[mat][k][n]   (3 x 128 x 2048)
// ---------------------------------------------------------------------------
__global__ __launch_bounds__(256) void wtrans_kernel(
    const float* __restrict__ Wq, const float* __restrict__ Wk,
    const float* __restrict__ Wv, __bf16* __restrict__ wt) {
  const int blk = blockIdx.x;        // 0..383
  const int mat = blk >> 7;          // 0..2
  const int n   = blk & 127;
  const float* W = (mat == 0) ? Wq : (mat == 1) ? Wk : Wv;
  __bf16* dst = wt + ((size_t)mat * HEAD_ + n) * EMB_;
  const int k0 = threadIdx.x * 8;
  __bf16 tmp[8];
#pragma unroll
  for (int i = 0; i < 8; ++i)
    tmp[i] = (__bf16)W[(size_t)(k0 + i) * HEAD_ + n];
  *reinterpret_cast<v8bf*>(dst + k0) = *reinterpret_cast<v8bf*>(tmp);
}

// ---------------------------------------------------------------------------
// Kernel 1: q/k/v projection (WMMA over K=2048) + bias + RoPE.
//   grid = B*S/16 workgroups of 256 threads (8 waves).
//   Wave w owns output columns [16w,16w+16) for all three matrices.
//   Outputs: qg,kg bf16 row-major [B*S][128]; vtg bf16 transposed [B][128][S].
// ---------------------------------------------------------------------------
__global__ __launch_bounds__(256) void proj_rope_kernel(
    const float* __restrict__ x,
    const float* __restrict__ bq, const float* __restrict__ bk,
    const float* __restrict__ bv,
    const __bf16* __restrict__ wt,
    __bf16* __restrict__ qg, __bf16* __restrict__ kg,
    __bf16* __restrict__ vtg) {
  __shared__ __align__(16) __bf16 xs[16 * 256];       // 8 KB  x-tile stage
  __shared__ __align__(16) float  qkbuf[2][16 * HEAD_]; // 16 KB rope stage

  const int wg   = blockIdx.x;            // 0..1023
  const int tid  = threadIdx.x;
  const int wave = tid >> 5;              // 0..7 -> column block
  const int lane = tid & 31;
  const int h    = lane >> 4;             // half-wave select
  const int n16  = lane & 15;
  const int col  = wave * 16 + n16;       // 0..127

  const __bf16* wtq = wt;
  const __bf16* wtk = wt + (size_t)HEAD_ * EMB_;
  const __bf16* wtv = wt + (size_t)2 * HEAD_ * EMB_;

  const size_t row0 = (size_t)wg * 16;    // global row of tile start

  v8f accQ = {}, accK = {}, accV = {};

  const int cm = tid >> 4;                // coop-stage row   0..15
  const int cc = (tid & 15) * 16;         // coop-stage col   0..240

  for (int kc = 0; kc < EMB_; kc += 256) {
    // ---- cooperative stage: 16x256 f32 tile of x -> bf16 in LDS ----
    {
      const float* src = x + (row0 + cm) * EMB_ + kc + cc;
      __bf16 tmp[16];
#pragma unroll
      for (int i = 0; i < 4; ++i) {
        v4f f = *reinterpret_cast<const v4f*>(src + 4 * i);
#pragma unroll
        for (int j = 0; j < 4; ++j) tmp[4 * i + j] = (__bf16)f[j];
      }
      *reinterpret_cast<v8bf*>(&xs[cm * 256 + cc])     = *reinterpret_cast<v8bf*>(tmp);
      *reinterpret_cast<v8bf*>(&xs[cm * 256 + cc + 8]) = *reinterpret_cast<v8bf*>(tmp + 8);
    }
    __syncthreads();

#pragma unroll
    for (int kb = 0; kb < 8; ++kb) {
      const int ck = kb * 32;
      // A fragment (16x32): lane = row n16, K = 16h+0..15 contiguous
      v16bf a = join16(
          *reinterpret_cast<const v8bf*>(&xs[n16 * 256 + ck + 16 * h]),
          *reinterpret_cast<const v8bf*>(&xs[n16 * 256 + ck + 16 * h + 8]));
      // B fragments (32x16): lane = col, K = 16h+0..15 contiguous in Wt rows
      const size_t wb = (size_t)col * EMB_ + kc + ck + 16 * h;
      v16bf fq = join16(*reinterpret_cast<const v8bf*>(wtq + wb),
                        *reinterpret_cast<const v8bf*>(wtq + wb + 8));
      v16bf fk = join16(*reinterpret_cast<const v8bf*>(wtk + wb),
                        *reinterpret_cast<const v8bf*>(wtk + wb + 8));
      v16bf fv = join16(*reinterpret_cast<const v8bf*>(wtv + wb),
                        *reinterpret_cast<const v8bf*>(wtv + wb + 8));
      accQ = __builtin_amdgcn_wmma_f32_16x16x32_bf16(false, a, false, fq,
                                                     (short)0, accQ, false, false);
      accK = __builtin_amdgcn_wmma_f32_16x16x32_bf16(false, a, false, fk,
                                                     (short)0, accK, false, false);
      accV = __builtin_amdgcn_wmma_f32_16x16x32_bf16(false, a, false, fv,
                                                     (short)0, accV, false, false);
    }
    __syncthreads();
  }

  // ---- bias; q/k to LDS (RoPE pairs (i, i+64) span waves) ----
  const float bqv = bq[col], bkv = bk[col], bvv = bv[col];
#pragma unroll
  for (int j = 0; j < 8; ++j) {
    qkbuf[0][(j + 8 * h) * HEAD_ + col] = accQ[j] + bqv;
    qkbuf[1][(j + 8 * h) * HEAD_ + col] = accK[j] + bkv;
  }

  // ---- v: store transposed bf16 directly (rows j+8h are contiguous in s) ----
  {
    __bf16 vp[8];
#pragma unroll
    for (int j = 0; j < 8; ++j) vp[j] = (__bf16)(accV[j] + bvv);
    const int bidx = wg >> 7;            // wg / (S/16)
    const int s0   = (wg & 127) * 16;
    __bf16* dst = vtg + ((size_t)bidx * HEAD_ + col) * S_ + s0 + 8 * h;
    *reinterpret_cast<v8bf*>(dst) = *reinterpret_cast<v8bf*>(vp);
  }
  __syncthreads();

  // ---- RoPE + bf16 store of q,k (thread t -> row t/16, cols (t%16)*8..+7) ----
  {
    const int m  = tid >> 4;
    const int c0 = (tid & 15) * 8;
    const int s0 = (wg & 127) * 16;
    const float spos = (float)(s0 + m);
    __bf16 qo[8], ko[8];
#pragma unroll
    for (int i = 0; i < 8; ++i) {
      const int c  = c0 + i;
      const int hi = (c >= 64);
      const int ci = c & 63;
      // inv_freq = 10000^(-ci/64) = 2^(-ci * log2(10000)/64)
      const float invf = exp2f(-(float)ci * (13.287712379549449f / 64.0f));
      const float ang  = spos * invf;
      const float ca = __cosf(ang), sa = __sinf(ang);
      const float q1 = qkbuf[0][m * HEAD_ + ci];
      const float q2 = qkbuf[0][m * HEAD_ + 64 + ci];
      const float k1 = qkbuf[1][m * HEAD_ + ci];
      const float k2 = qkbuf[1][m * HEAD_ + 64 + ci];
      qo[i] = (__bf16)(hi ? (q1 * sa + q2 * ca) : (q1 * ca - q2 * sa));
      ko[i] = (__bf16)(hi ? (k1 * sa + k2 * ca) : (k1 * ca - k2 * sa));
    }
    *reinterpret_cast<v8bf*>(qg + (row0 + m) * HEAD_ + c0) = *reinterpret_cast<v8bf*>(qo);
    *reinterpret_cast<v8bf*>(kg + (row0 + m) * HEAD_ + c0) = *reinterpret_cast<v8bf*>(ko);
  }
}

// ---------------------------------------------------------------------------
// Kernel 2: causal flash attention.
//   grid = B * S/64 workgroups of 128 threads; 4 INDEPENDENT waves
//   (no workgroup barrier -> per-wave causal loop bounds are safe).
//   Each wave: 16 q rows, kv steps of 32, online softmax, 16x128 f32 O acc.
// ---------------------------------------------------------------------------
__global__ __launch_bounds__(128) void attn_kernel(
    const __bf16* __restrict__ qg, const __bf16* __restrict__ kg,
    const __bf16* __restrict__ vtg, float* __restrict__ out) {
  __shared__ __align__(16) __bf16 pbuf[4][16 * 32];   // per-wave P stage, 4 KB

  const int bidx = blockIdx.x >> 5;     // batch (32 q-blocks per batch)
  const int qb   = blockIdx.x & 31;
  const int tid  = threadIdx.x;
  const int wave = tid >> 5;
  const int lane = tid & 31;
  const int h    = lane >> 4;
  const int n16  = lane & 15;

  const int   sq0   = qb * 64 + wave * 16;        // first q row of this wave
  const float scale = 0.08838834764831845f;       // 1/sqrt(128)
  const float L2E   = 1.4426950408889634f;

  // Q A-fragments: 16x128 bf16 held in registers (4 fragments of K=32)
  v16bf qa[4];
  {
    const __bf16* qrow = qg + ((size_t)bidx * S_ + sq0 + n16) * HEAD_;
#pragma unroll
    for (int f = 0; f < 4; ++f)
      qa[f] = join16(*reinterpret_cast<const v8bf*>(qrow + 32 * f + 16 * h),
                     *reinterpret_cast<const v8bf*>(qrow + 32 * f + 16 * h + 8));
  }

  v8f o[8];
#pragma unroll
  for (int t = 0; t < 8; ++t) o[t] = (v8f){};
  float mcur[8], lsum[8];
#pragma unroll
  for (int j = 0; j < 8; ++j) { mcur[j] = -__builtin_inff(); lsum[j] = 0.f; }

  const int nsteps = sq0 / 32 + 1;      // kv covers [0, sq0+16)
  for (int it = 0; it < nsteps; ++it) {
    const int kv0 = it * 32;

    // ---- scores: two 16x16 tiles, K=128 reduction (8 WMMAs) ----
    v8f sA = (v8f){}, sB = (v8f){};
    {
      const __bf16* kr0 = kg + ((size_t)bidx * S_ + kv0 + n16) * HEAD_;
      const __bf16* kr1 = kr0 + (size_t)16 * HEAD_;
#pragma unroll
      for (int f = 0; f < 4; ++f) {
        v16bf b0 = join16(*reinterpret_cast<const v8bf*>(kr0 + 32 * f + 16 * h),
                          *reinterpret_cast<const v8bf*>(kr0 + 32 * f + 16 * h + 8));
        v16bf b1 = join16(*reinterpret_cast<const v8bf*>(kr1 + 32 * f + 16 * h),
                          *reinterpret_cast<const v8bf*>(kr1 + 32 * f + 16 * h + 8));
        sA = __builtin_amdgcn_wmma_f32_16x16x32_bf16(false, qa[f], false, b0,
                                                     (short)0, sA, false, false);
        sB = __builtin_amdgcn_wmma_f32_16x16x32_bf16(false, qa[f], false, b1,
                                                     (short)0, sB, false, false);
      }
    }

    // ---- causal mask + online softmax (row = j + 8h, col = n16) ----
#pragma unroll
    for (int j = 0; j < 8; ++j) {
      const int sq = sq0 + j + 8 * h;
      float a0 = sA[j] * scale;
      float a1 = sB[j] * scale;
      if (kv0 + n16 > sq)      a0 = -__builtin_inff();
      if (kv0 + 16 + n16 > sq) a1 = -__builtin_inff();
      float mx = fmaxf(a0, a1);
#pragma unroll
      for (int d = 1; d < 16; d <<= 1) mx = fmaxf(mx, __shfl_xor(mx, d, 32));
      const float mnew  = fmaxf(mcur[j], mx);
      const float alpha = exp2f((mcur[j] - mnew) * L2E);
      const float p0 = exp2f((a0 - mnew) * L2E);
      const float p1 = exp2f((a1 - mnew) * L2E);
      float rs = p0 + p1;
#pragma unroll
      for (int d = 1; d < 16; d <<= 1) rs += __shfl_xor(rs, d, 32);
      lsum[j] = lsum[j] * alpha + rs;
      mcur[j] = mnew;
#pragma unroll
      for (int t = 0; t < 8; ++t) o[t][j] *= alpha;
      pbuf[wave][(j + 8 * h) * 32 + n16]      = (__bf16)p0;
      pbuf[wave][(j + 8 * h) * 32 + 16 + n16] = (__bf16)p1;
    }
    // D-layout -> A-layout through wave-private LDS; lanes are lockstep,
    // so a dscnt wait (not a workgroup barrier) orders store->load.
    asm volatile("s_wait_dscnt 0" ::: "memory");
    v16bf pa = join16(
        *reinterpret_cast<const v8bf*>(&pbuf[wave][n16 * 32 + 16 * h]),
        *reinterpret_cast<const v8bf*>(&pbuf[wave][n16 * 32 + 16 * h + 8]));
    asm volatile("s_wait_dscnt 0" ::: "memory");

    // ---- P (16x32) x V (32x128): 8 WMMAs, B from transposed v ----
#pragma unroll
    for (int t = 0; t < 8; ++t) {
      const __bf16* vrow =
          vtg + ((size_t)bidx * HEAD_ + 16 * t + n16) * S_ + kv0 + 16 * h;
      v16bf bvf = join16(*reinterpret_cast<const v8bf*>(vrow),
                         *reinterpret_cast<const v8bf*>(vrow + 8));
      o[t] = __builtin_amdgcn_wmma_f32_16x16x32_bf16(false, pa, false, bvf,
                                                     (short)0, o[t], false, false);
    }
  }

  // ---- normalize + f32 store ----
#pragma unroll
  for (int j = 0; j < 8; ++j) {
    const float inv = 1.0f / lsum[j];
    float* orow = out + ((size_t)bidx * S_ + sq0 + j + 8 * h) * HEAD_ + n16;
#pragma unroll
    for (int t = 0; t < 8; ++t) orow[16 * t] = o[t][j] * inv;
  }
}

// ---------------------------------------------------------------------------
// Host launcher
// ---------------------------------------------------------------------------
extern "C" void kernel_launch(void* const* d_in, const int* in_sizes, int n_in,
                              void* d_out, int out_size, void* d_ws, size_t ws_size,
                              hipStream_t stream) {
  const float* x  = (const float*)d_in[0];
  const float* Wq = (const float*)d_in[1];
  const float* bq = (const float*)d_in[2];
  const float* Wk = (const float*)d_in[3];
  const float* bk = (const float*)d_in[4];
  const float* Wv = (const float*)d_in[5];
  const float* bv = (const float*)d_in[6];
  float* out = (float*)d_out;

  // Workspace layout (bf16): Wt[3][128][2048] | q[B*S][128] | k[B*S][128] | vT[B][128][S]
  __bf16* wt  = (__bf16*)d_ws;
  __bf16* qg  = wt + (size_t)3 * HEAD_ * EMB_;
  __bf16* kg  = qg + (size_t)B_ * S_ * HEAD_;
  __bf16* vtg = kg + (size_t)B_ * S_ * HEAD_;

  wtrans_kernel<<<3 * HEAD_, 256, 0, stream>>>(Wq, Wk, Wv, wt);
  proj_rope_kernel<<<(B_ * S_) / 16, 256, 0, stream>>>(x, bq, bk, bv, wt,
                                                       qg, kg, vtg);
  attn_kernel<<<B_ * (S_ / 64), 128, 0, stream>>>(qg, kg, vtg, out);
}
